// RNN_70755291234821
// MI455X (gfx1250) — matrix operand order repaired and checked by reading, not compiled, not run on hardware
//
#include <hip/hip_runtime.h>
#include <math.h>

// ---------------- problem constants ----------------
#define HID    512
#define INP    64
#define OUTD   64
#define BATCH  256
#define TSTEPS 1024
// K is tiled in chunks of 32 (bf16 WMMA K): 16 chunks for W_rec (K=512)
// plus 2 chunks for W_in (K=64) -> fused [r | u] GEMM with K=576.
#define NKC_H   16
#define NKC     18
#define NTILES_H 32           // 512/16 output column tiles of r_new
#define NTILES_O 4            // 64/16 output column tiles of o
#define ROWSTR  592           // LDS A-buffer row stride (576 used + pad), 16B aligned

typedef __attribute__((ext_vector_type(16))) __bf16 v16bf;
typedef __attribute__((ext_vector_type(8)))  __bf16 bf16x8;
typedef __attribute__((ext_vector_type(8)))  float  v8f;

union AFrag { v16bf v; bf16x8 g[2]; };

#define PW_ELTS (NTILES_H * NKC * 512)     // packed [W_rec|W_in] bf16 elements (294912)
#define PO_ELTS (NTILES_O * NKC_H * 512)   // packed W_out bf16 elements (32768)

__device__ __forceinline__ v8f wmma_bf16(v16bf a, v16bf b, v8f c) {
  // D = A(16x32 bf16) * B(32x16 bf16) + C(16x16 f32)
  return __builtin_amdgcn_wmma_f32_16x16x32_bf16(
      /*neg_a=*/false, a, /*neg_b=*/false, b,
      /*c_mod=*/(short)0, c, /*reuse_a=*/false, /*reuse_b=*/false);
}

__device__ __forceinline__ float fast_tanh(float x) {
#if __has_builtin(__builtin_amdgcn_tanhf)
  return __builtin_amdgcn_tanhf(x);   // gfx1250 v_tanh_f32 (TRANS pipe)
#else
  return tanhf(x);
#endif
}

// ---------------------------------------------------------------------------
// One-time repack of f32 weights into bf16 WMMA B-fragment layout.
// B-matrix 32x16 bf16 layout used here (not the ISA table order): lane = N + 16*(K>=16),
// each lane holds 16 *contiguous* K values -> fragment = 32B/lane, 1KB/wave,
// fully coalesced global loads in the hot loop.
// ---------------------------------------------------------------------------
__global__ void rnn_pack_weights(const float* __restrict__ Wrec,
                                 const float* __restrict__ Win,
                                 const float* __restrict__ Wout,
                                 __bf16* __restrict__ pW,
                                 __bf16* __restrict__ pO) {
  int idx = blockIdx.x * blockDim.x + threadIdx.x;
  if (idx < PW_ELTS) {
    int j     = idx & 15;
    int lane  = (idx >> 4) & 31;
    int frag  = idx >> 9;             // ntile*NKC + kc
    int kc    = frag % NKC;
    int ntile = frag / NKC;
    int h     = ntile * 16 + (lane & 15);
    int kk    = (lane >> 4) * 16 + j;
    float v;
    if (kc < NKC_H) v = Wrec[(size_t)h * HID + kc * 32 + kk];
    else            v = Win [(size_t)h * INP + (kc - NKC_H) * 32 + kk];
    pW[idx] = (__bf16)v;
  } else if (idx < PW_ELTS + PO_ELTS) {
    int t     = idx - PW_ELTS;
    int j     = t & 15;
    int lane  = (t >> 4) & 31;
    int frag  = t >> 9;
    int kc    = frag & 15;
    int otile = frag >> 4;
    pO[t] = (__bf16)Wout[(size_t)(otile * 16 + (lane & 15)) * HID +
                         kc * 32 + (lane >> 4) * 16 + j];
  }
}

// ---------------------------------------------------------------------------
// Persistent recurrence kernel: one workgroup (8 wave32 = 256 thr) per 16-row
// batch tile; 16 workgroups total (batch rows are independent -> zero
// cross-WGP sync). State r (16x512 f32) lives in VGPRs as D-tile fragments
// (wave w owns column strips w*4..w*4+3). LDS holds a double-buffered bf16
// copy of [r_t | u_t] in exact WMMA A-fragment order so every wave feeds its
// WMMAs with two ds_load_b128 per K-chunk; one s_barrier per timestep.
// o_{t-1} = r_t @ W_out^T is folded INTO the main K-loop on waves 0-3: it
// reuses each A fragment right as it is loaded. (Splitting it into a separate
// pass makes the scheduler hoist all 36 A ds_loads and spill B fragments to
// scratch; keep it interleaved.)
// ---------------------------------------------------------------------------
__global__ __launch_bounds__(256, 1)
void rnn_persistent(const float* __restrict__ U,
                    const float* __restrict__ r0,
                    const __bf16* __restrict__ pW,
                    const __bf16* __restrict__ pO,
                    float* __restrict__ R,
                    float* __restrict__ O) {
  __shared__ __attribute__((aligned(16))) __bf16 Abuf[2][16][ROWSTR];

  const int tid  = threadIdx.x;
  const int wave = tid >> 5;
  const int lane = tid & 31;
  const int nlo  = lane & 15;    // A-frag: M row; B/D-frag: N column
  const int hD   = lane >> 4;    // lane-half selector
  const int h8   = hD * 8;
  const int b0   = blockIdx.x * 16;

  v8f rstate[4];

  // ---- init: r0 -> registers + R[:,0,:] + Abuf[0]; u_0 -> Abuf[0] ----
  #pragma unroll
  for (int s = 0; s < 4; ++s) {
    const int hb = (wave * 4 + s) * 16 + nlo;
    v8f rs = {};
    #pragma unroll
    for (int e = 0; e < 8; ++e) {
      const int m = e + 8 * hD;                       // D-tile row mapping
      float r = r0[(size_t)(b0 + m) * HID + hb];
      rs[e] = r;
      __builtin_nontemporal_store(r, &R[((size_t)(b0 + m) * (TSTEPS + 1)) * HID + hb]);
      Abuf[0][m][hb] = (__bf16)r;
    }
    rstate[s] = rs;
  }
  {
    const int row = tid >> 4;
    const int c0  = (tid & 15) * 4;
    const float4 u = *(const float4*)&U[((size_t)(b0 + row) * TSTEPS + 0) * INP + c0];
    Abuf[0][row][HID + c0 + 0] = (__bf16)u.x;
    Abuf[0][row][HID + c0 + 1] = (__bf16)u.y;
    Abuf[0][row][HID + c0 + 2] = (__bf16)u.z;
    Abuf[0][row][HID + c0 + 3] = (__bf16)u.w;
  }
  __syncthreads();

  for (int t = 0; t < TSTEPS; ++t) {
    const int p  = t & 1;       // read buffer  (holds r_t, u_t)
    const int wb = p ^ 1;       // write buffer (gets  r_{t+1}, u_{t+1})
    const bool doO = (wave < NTILES_O) && (t > 0);

    v8f acc[4]; v8f accO = {};
    #pragma unroll
    for (int s = 0; s < 4; ++s) { v8f z = {}; acc[s] = z; }

    #pragma unroll
    for (int kc = 0; kc < NKC; ++kc) {
      // A fragment (shared across all column strips of this wave):
      // A 16x32 bf16 layout: lane m=nlo, half hD -> K
      // [kc*32 + 16*hD .. +15] (contiguous, matches packed B)
      AFrag a;
      a.g[0] = *(const bf16x8*)&Abuf[p][nlo][kc * 32 + 2 * h8];
      a.g[1] = *(const bf16x8*)&Abuf[p][nlo][kc * 32 + 2 * h8 + 8];

      const __bf16* wbase =
          pW + ((size_t)((wave * 4) * NKC + kc) * 32 + lane) * 16;
      #pragma unroll
      for (int s = 0; s < 4; ++s) {
        AFrag b;
        const __bf16* bp = wbase + (size_t)s * (NKC * 512);
        b.g[0] = *(const bf16x8*)bp;
        b.g[1] = *(const bf16x8*)(bp + 8);
        acc[s] = wmma_bf16(a.v, b.v, acc[s]);
      }
      if (doO && kc < NKC_H) {   // o_{t-1} = r_t @ W_out^T, reuses A frags
        AFrag b;
        const __bf16* op = pO + ((size_t)(wave * NKC_H + kc) * 32 + lane) * 16;
        b.g[0] = *(const bf16x8*)op;
        b.g[1] = *(const bf16x8*)(op + 8);
        accO = wmma_bf16(a.v, b.v, accO);
      }
    }

    // r_{t+1} = 0.5*(r_t + tanh(act)); stream R (NT), refill Abuf[wb]
    #pragma unroll
    for (int s = 0; s < 4; ++s) {
      const int hb = (wave * 4 + s) * 16 + nlo;
      v8f rs = rstate[s];
      v8f av = acc[s];
      #pragma unroll
      for (int e = 0; e < 8; ++e) {
        const int m = e + 8 * hD;
        float rn = 0.5f * (rs[e] + fast_tanh(av[e]));
        rs[e] = rn;
        __builtin_nontemporal_store(
            rn, &R[((size_t)(b0 + m) * (TSTEPS + 1) + (t + 1)) * HID + hb]);
        Abuf[wb][m][hb] = (__bf16)rn;
      }
      rstate[s] = rs;
    }
    if (doO) {
      #pragma unroll
      for (int e = 0; e < 8; ++e) {
        const int m = e + 8 * hD;
        __builtin_nontemporal_store(
            accO[e],
            &O[((size_t)(b0 + m) * TSTEPS + (t - 1)) * OUTD + wave * 16 + nlo]);
      }
    }
    if (t + 1 < TSTEPS) {   // stage u_{t+1} into the write buffer
      const int row = tid >> 4;
      const int c0  = (tid & 15) * 4;
      const float4 u =
          *(const float4*)&U[((size_t)(b0 + row) * TSTEPS + (t + 1)) * INP + c0];
      Abuf[wb][row][HID + c0 + 0] = (__bf16)u.x;
      Abuf[wb][row][HID + c0 + 1] = (__bf16)u.y;
      Abuf[wb][row][HID + c0 + 2] = (__bf16)u.z;
      Abuf[wb][row][HID + c0 + 3] = (__bf16)u.w;
    }
    __syncthreads();   // single barrier per step (double-buffered Abuf)
  }

  // epilogue: o_{T-1} = r_T @ W_out^T, r_T sits in Abuf[TSTEPS & 1]
  if (wave < NTILES_O) {
    v8f accO = {};
    #pragma unroll
    for (int kc = 0; kc < NKC_H; ++kc) {
      AFrag a, b;
      a.g[0] = *(const bf16x8*)&Abuf[TSTEPS & 1][nlo][kc * 32 + 2 * h8];
      a.g[1] = *(const bf16x8*)&Abuf[TSTEPS & 1][nlo][kc * 32 + 2 * h8 + 8];
      const __bf16* op = pO + ((size_t)(wave * NKC_H + kc) * 32 + lane) * 16;
      b.g[0] = *(const bf16x8*)op;
      b.g[1] = *(const bf16x8*)(op + 8);
      accO = wmma_bf16(a.v, b.v, accO);
    }
    #pragma unroll
    for (int e = 0; e < 8; ++e) {
      const int m = e + 8 * hD;
      __builtin_nontemporal_store(
          accO[e],
          &O[((size_t)(b0 + m) * TSTEPS + (TSTEPS - 1)) * OUTD + wave * 16 + nlo]);
    }
  }
}

extern "C" void kernel_launch(void* const* d_in, const int* in_sizes, int n_in,
                              void* d_out, int out_size, void* d_ws, size_t ws_size,
                              hipStream_t stream) {
  (void)in_sizes; (void)n_in; (void)out_size; (void)ws_size;
  const float* U    = (const float*)d_in[0];
  const float* r0   = (const float*)d_in[1];
  const float* Wrec = (const float*)d_in[2];
  const float* Win  = (const float*)d_in[3];
  const float* Wout = (const float*)d_in[4];

  float* R = (float*)d_out;                                   // (B, T+1, H)
  float* O = R + (size_t)BATCH * (TSTEPS + 1) * HID;          // (B, T, OUT)

  __bf16* pW = (__bf16*)d_ws;            // 294912 bf16 (~576 KB)
  __bf16* pO = pW + PW_ELTS;             //  32768 bf16 (~64 KB)

  const int packElems = PW_ELTS + PO_ELTS;
  rnn_pack_weights<<<(packElems + 255) / 256, 256, 0, stream>>>(
      Wrec, Win, Wout, pW, pO);
  rnn_persistent<<<BATCH / 16, 256, 0, stream>>>(U, r0, pW, pO, R, O);
}
